// Attention_34978213658826
// MI455X (gfx1250) — matrix operand order repaired and checked by reading, not compiled, not run on hardware
//
#include <hip/hip_runtime.h>
#include <hip/hip_bf16.h>

// ---------------------------------------------------------------------------
// Causal MHA for MI455X (gfx1250, wave32, WMMA).
// B=2, L=2048, N_HEADS=16, D_HEAD=64, D_MODEL=1024.
// All matmuls use v_wmma_f32_16x16x32_bf16 with f32 accumulation. Working set
// (~50MB bf16 staging) lives in L2 (192MB); kernels are matrix-pipe bound.
// GEMM K-loops are manually ping-pong unrolled (2 fragment sets, K step 64)
// so prefetched loads land directly in their consuming registers: no
// rotate-moves, and every 8-WMMA block covers the other set's load latency.
// ---------------------------------------------------------------------------

#define DM    1024
#define NH    16
#define DH    64
#define BATCH 2
#define SEQ   2048

typedef __attribute__((ext_vector_type(16))) __bf16 v16bf;
typedef __attribute__((ext_vector_type(8)))  float  v8f;

struct U32B { uint4 a, b; };   // 32 bytes

static __device__ __forceinline__ v16bf ldfrag1(const __bf16* p) {
  // 16 contiguous bf16 (32B) — B-fragment: lane ln -> col, K = half*16 + e
  U32B u;
  u.a = *reinterpret_cast<const uint4*>(p);
  u.b = *reinterpret_cast<const uint4*>(p + 8);
  return __builtin_bit_cast(v16bf, u);
}

static __device__ __forceinline__ v16bf ldfrag2(const __bf16* p0, const __bf16* p1) {
  // A-fragment: two contiguous 8-bf16 chunks, K = kb..kb+7 and kb+8..kb+15
  U32B u;
  u.a = *reinterpret_cast<const uint4*>(p0);
  u.b = *reinterpret_cast<const uint4*>(p1);
  return __builtin_bit_cast(v16bf, u);
}

static __device__ __forceinline__ __bf16 f2bf(float f) {
  union { float f; unsigned u; } a; a.f = f;
  unsigned r = a.u + 0x7FFFu + ((a.u >> 16) & 1u);   // round-to-nearest-even
  unsigned short h = (unsigned short)(r >> 16);
  return __builtin_bit_cast(__bf16, h);
}

static __device__ __forceinline__ v8f zero8() {
  v8f z;
#pragma unroll
  for (int i = 0; i < 8; ++i) z[i] = 0.f;
  return z;
}

static __device__ __forceinline__ v8f wmma_bf16(v16bf a, v16bf b, v8f c) {
  return __builtin_amdgcn_wmma_f32_16x16x32_bf16(false, a, false, b, (short)0, c,
                                                 false, false);
}

// ---------------------------------------------------------------------------
// Conversion kernels (one-time, tiny traffic)
// ---------------------------------------------------------------------------
__global__ void cvt_f32_bf16(const float* __restrict__ src,
                             __bf16* __restrict__ dst, int n) {
  int i = blockIdx.x * blockDim.x + threadIdx.x;
  if (i < n) dst[i] = f2bf(src[i]);
}

// W[n][m][h] -> Wt[n][h][m]
__global__ void cvt_w_headT(const float* __restrict__ src,
                            __bf16* __restrict__ dst) {
  int i = blockIdx.x * blockDim.x + threadIdx.x;   // over NH*DM*DH
  int n = i >> 16;            // DM*DH = 65536
  int rem = i & 65535;
  int m = rem >> 6, h = rem & 63;
  dst[(n << 16) + (h << 10) + m] = f2bf(src[i]);
}

// W_O[n][h][m] (== [k][m], k=n*64+h) -> WoT[m][k]
__global__ void cvt_woT(const float* __restrict__ src,
                        __bf16* __restrict__ dst) {
  int i = blockIdx.x * blockDim.x + threadIdx.x;   // over DM*DM (dst index)
  int m = i >> 10, k = i & 1023;
  dst[i] = f2bf(src[k * DM + m]);
}

// ---------------------------------------------------------------------------
// QKV projection: per (b,n,{q,k,v}), C(L x 64) = X(L x 1024) * W(1024 x 64).
// One wave computes a 32x64 tile (2 M-subtiles sharing B-fragments).
// K loop: step 64, two ping-pong fragment sets (no rotate copies).
// Q,K stored [b][n][l][h]; V stored transposed [b][n][h][l] for the PV stage.
// ---------------------------------------------------------------------------
__global__ __launch_bounds__(128) void qkv_gemm(
    const __bf16* __restrict__ Xb,
    const __bf16* __restrict__ WqT, const __bf16* __restrict__ WkT,
    const __bf16* __restrict__ WvT,
    const float* __restrict__ bq, const float* __restrict__ bk,
    const float* __restrict__ bv,
    __bf16* __restrict__ Qo, __bf16* __restrict__ Ko,
    __bf16* __restrict__ Vo) {
  const int wave = threadIdx.x >> 5;
  const int lane = threadIdx.x & 31;
  const int ln = lane & 15, half = lane >> 4;
  const int bn = blockIdx.y;              // b*16 + n
  const int n = bn & 15;
  const int b = bn >> 4;
  const int m0 = (blockIdx.x * 4 + wave) * 32;

  const __bf16* Wt; const float* bias; __bf16* dst; bool vstore = false;
  if (blockIdx.z == 0)      { Wt = WqT; bias = bq; dst = Qo; }
  else if (blockIdx.z == 1) { Wt = WkT; bias = bk; dst = Ko; }
  else                      { Wt = WvT; bias = bv; dst = Vo; vstore = true; }

  const __bf16* x0p = Xb + ((size_t)b * SEQ + m0 + ln) * DM + half * 16;
  const __bf16* x1p = x0p + (size_t)16 * DM;
  const __bf16* wr  = Wt + ((size_t)n * DH + ln) * DM + half * 16;

  v8f acc[2][4];
#pragma unroll
  for (int mt = 0; mt < 2; ++mt)
#pragma unroll
    for (int nt = 0; nt < 4; ++nt) acc[mt][nt] = zero8();

  // set 0 primed at kk = 0
  v16bf a0c = ldfrag2(x0p, x0p + 8);
  v16bf a1c = ldfrag2(x1p, x1p + 8);
  v16bf bfc[4];
#pragma unroll
  for (int nt = 0; nt < 4; ++nt) bfc[nt] = ldfrag1(wr + (size_t)nt * 16 * DM);

  for (int kk = 0; kk < DM; kk += 64) {
    // prefetch set 1 at kk+32 (always in range: kk+32 <= DM-32)
    const int k1 = kk + 32;
    v16bf a0n = ldfrag2(x0p + k1, x0p + k1 + 8);
    v16bf a1n = ldfrag2(x1p + k1, x1p + k1 + 8);
    v16bf bfn[4];
#pragma unroll
    for (int nt = 0; nt < 4; ++nt)
      bfn[nt] = ldfrag1(wr + (size_t)nt * 16 * DM + k1);

#pragma unroll
    for (int nt = 0; nt < 4; ++nt) {
      acc[0][nt] = wmma_bf16(a0c, bfc[nt], acc[0][nt]);
      acc[1][nt] = wmma_bf16(a1c, bfc[nt], acc[1][nt]);
    }

    // reload set 0 at kk+64 (skipped only on final iteration)
    const int k2 = kk + 64;
    if (k2 < DM) {
      a0c = ldfrag2(x0p + k2, x0p + k2 + 8);
      a1c = ldfrag2(x1p + k2, x1p + k2 + 8);
#pragma unroll
      for (int nt = 0; nt < 4; ++nt)
        bfc[nt] = ldfrag1(wr + (size_t)nt * 16 * DM + k2);
    }

#pragma unroll
    for (int nt = 0; nt < 4; ++nt) {
      acc[0][nt] = wmma_bf16(a0n, bfn[nt], acc[0][nt]);
      acc[1][nt] = wmma_bf16(a1n, bfn[nt], acc[1][nt]);
    }
  }

#pragma unroll
  for (int mt = 0; mt < 2; ++mt)
#pragma unroll
    for (int nt = 0; nt < 4; ++nt) {
      float bs = bias[n * DH + nt * 16 + ln];
#pragma unroll
      for (int r = 0; r < 8; ++r) {
        float val = acc[mt][nt][r] + bs;
        int mrow = m0 + mt * 16 + half * 8 + r;
        if (!vstore)
          dst[((size_t)bn * SEQ + mrow) * DH + nt * 16 + ln] = f2bf(val);
        else
          dst[((size_t)bn * DH + nt * 16 + ln) * SEQ + mrow] = f2bf(val);
      }
    }
}

// ---------------------------------------------------------------------------
// Flash-style causal attention. One wave per 16-row Q tile; 8 waves / block.
// K processed in 32-wide tiles. P is ping-pong double-buffered in LDS (one
// __syncthreads per tile; block-uniform trip count; waves past their diagonal
// run fully-masked iterations, p == 0). K fragments for the next tile are
// prefetched unconditionally (clamped address) right after consumption, so
// the loads retarget the same carried registers with no copies; softmax +
// barrier + PV cover their latency. V fragments load before the softmax.
// ---------------------------------------------------------------------------
__global__ __launch_bounds__(256) void attn_kernel(
    const __bf16* __restrict__ Q, const __bf16* __restrict__ K,
    const __bf16* __restrict__ Vt, __bf16* __restrict__ Z) {
  __shared__ __align__(16) __bf16 plds[8 * 2 * 512];   // 8 waves * 2 buffers * 1KB

  const int wave = threadIdx.x >> 5;
  const int lane = threadIdx.x & 31;
  const int ln = lane & 15, half = lane >> 4;
  const int bn = blockIdx.y;               // b*16 + n
  const int n = bn & 15, b = bn >> 4;
  const int q0 = blockIdx.x * 128 + wave * 16;

  const __bf16* qrow = Q + ((size_t)bn * SEQ + q0 + ln) * DH;
  v16bf a0 = ldfrag2(qrow + half * 16,      qrow + half * 16 + 8);       // h 0..31
  v16bf a1 = ldfrag2(qrow + 32 + half * 16, qrow + 32 + half * 16 + 8);  // h 32..63

  float mrow[8], lrow[8];
#pragma unroll
  for (int r = 0; r < 8; ++r) { mrow[r] = -1e30f; lrow[r] = 0.f; }
  v8f zacc[4];
#pragma unroll
  for (int nt = 0; nt < 4; ++nt) zacc[nt] = zero8();

  __bf16* pl0 = plds + wave * 1024;
  const int jmax = blockIdx.x * 4 + 4;     // block-uniform trip count

  const __bf16* kbase = K + (size_t)bn * SEQ * DH;
  const __bf16* vbase = Vt + (size_t)bn * DH * SEQ;

  // prime K fragments for j = 0
  v16bf kf[4];
  kf[0] = ldfrag1(kbase + (size_t)ln * DH + half * 16);
  kf[1] = ldfrag1(kbase + (size_t)ln * DH + 32 + half * 16);
  kf[2] = ldfrag1(kbase + (size_t)(16 + ln) * DH + half * 16);
  kf[3] = ldfrag1(kbase + (size_t)(16 + ln) * DH + 32 + half * 16);

  for (int j = 0; j < jmax; ++j) {
    const int k0 = j * 32;

    // ---- V fragments for this tile (consumed only after softmax) ---------
    v16bf vf[4];
#pragma unroll
    for (int nt = 0; nt < 4; ++nt)
      vf[nt] = ldfrag1(vbase + (size_t)(nt * 16 + ln) * SEQ + k0 + half * 16);

    // ---- S = Q * K^T for two 16-wide key columns -------------------------
    v8f s0 = zero8(), s1 = zero8();
    s0 = wmma_bf16(a0, kf[0], s0); s0 = wmma_bf16(a1, kf[1], s0);
    s1 = wmma_bf16(a0, kf[2], s1); s1 = wmma_bf16(a1, kf[3], s1);

    // ---- prefetch next tile's K fragments (clamped, unconditional) -------
    {
      const int kn = (k0 + 32 <= SEQ - 32) ? (k0 + 32) : (SEQ - 32);
      const __bf16* kb = kbase + (size_t)kn * DH;
      kf[0] = ldfrag1(kb + (size_t)ln * DH + half * 16);
      kf[1] = ldfrag1(kb + (size_t)ln * DH + 32 + half * 16);
      kf[2] = ldfrag1(kb + (size_t)(16 + ln) * DH + half * 16);
      kf[3] = ldfrag1(kb + (size_t)(16 + ln) * DH + 32 + half * 16);
    }

    // ---- online softmax, write P into ping-pong LDS buffer ---------------
    __bf16* pl = pl0 + (j & 1) * 512;
    float corr[8];
#pragma unroll
    for (int r = 0; r < 8; ++r) {
      int qg = q0 + half * 8 + r;
      float v0 = s0[r] * 0.125f; if (k0 + ln > qg)      v0 = -1e30f;
      float v1 = s1[r] * 0.125f; if (k0 + 16 + ln > qg) v1 = -1e30f;
      float mx = fmaxf(v0, v1);
      mx = fmaxf(mx, __shfl_xor(mx, 1, 16));
      mx = fmaxf(mx, __shfl_xor(mx, 2, 16));
      mx = fmaxf(mx, __shfl_xor(mx, 4, 16));
      mx = fmaxf(mx, __shfl_xor(mx, 8, 16));
      float mn = fmaxf(mrow[r], mx);
      float c  = __expf(mrow[r] - mn);
      float p0 = __expf(v0 - mn);
      float p1 = __expf(v1 - mn);
      float sm = p0 + p1;
      sm += __shfl_xor(sm, 1, 16);
      sm += __shfl_xor(sm, 2, 16);
      sm += __shfl_xor(sm, 4, 16);
      sm += __shfl_xor(sm, 8, 16);
      lrow[r] = lrow[r] * c + sm;
      mrow[r] = mn; corr[r] = c;
      pl[(half * 8 + r) * 32 + ln]      = f2bf(p0);
      pl[(half * 8 + r) * 32 + 16 + ln] = f2bf(p1);
    }
#pragma unroll
    for (int nt = 0; nt < 4; ++nt)
#pragma unroll
      for (int r = 0; r < 8; ++r) zacc[nt][r] *= corr[r];

    __syncthreads();   // P stores ordered before A-fragment reload
    v16bf pa = ldfrag2(pl + ln * 32 + half * 16, pl + ln * 32 + half * 16 + 8);

    // ---- Z += P * V -------------------------------------------------------
#pragma unroll
    for (int nt = 0; nt < 4; ++nt)
      zacc[nt] = wmma_bf16(pa, vf[nt], zacc[nt]);
  }

  float inv[8];
#pragma unroll
  for (int r = 0; r < 8; ++r) inv[r] = 1.0f / lrow[r];
#pragma unroll
  for (int nt = 0; nt < 4; ++nt)
#pragma unroll
    for (int r = 0; r < 8; ++r) {
      size_t o = ((size_t)b * SEQ + q0 + half * 8 + r) * DM + n * DH + nt * 16 + ln;
      Z[o] = f2bf(zacc[nt][r] * inv[r]);
    }
}

// ---------------------------------------------------------------------------
// Output projection: out(L x 1024) = Z(L x 1024) * W_O(1024 x 1024) + b_O.
// Same ping-pong pipelined 32x64-per-wave structure; stores f32 to d_out.
// ---------------------------------------------------------------------------
__global__ __launch_bounds__(128) void out_gemm(
    const __bf16* __restrict__ Zb, const __bf16* __restrict__ WoT,
    const float* __restrict__ bO, float* __restrict__ out) {
  const int wave = threadIdx.x >> 5;
  const int lane = threadIdx.x & 31;
  const int ln = lane & 15, half = lane >> 4;
  const int b = blockIdx.z;
  const int m0 = (blockIdx.x * 4 + wave) * 32;
  const int n0 = blockIdx.y * 64;

  const __bf16* x0p = Zb + ((size_t)b * SEQ + m0 + ln) * DM + half * 16;
  const __bf16* x1p = x0p + (size_t)16 * DM;
  const __bf16* wr  = WoT + (size_t)(n0 + ln) * DM + half * 16;

  v8f acc[2][4];
#pragma unroll
  for (int mt = 0; mt < 2; ++mt)
#pragma unroll
    for (int nt = 0; nt < 4; ++nt) acc[mt][nt] = zero8();

  v16bf a0c = ldfrag2(x0p, x0p + 8);
  v16bf a1c = ldfrag2(x1p, x1p + 8);
  v16bf bfc[4];
#pragma unroll
  for (int nt = 0; nt < 4; ++nt) bfc[nt] = ldfrag1(wr + (size_t)nt * 16 * DM);

  for (int kk = 0; kk < DM; kk += 64) {
    const int k1 = kk + 32;
    v16bf a0n = ldfrag2(x0p + k1, x0p + k1 + 8);
    v16bf a1n = ldfrag2(x1p + k1, x1p + k1 + 8);
    v16bf bfn[4];
#pragma unroll
    for (int nt = 0; nt < 4; ++nt)
      bfn[nt] = ldfrag1(wr + (size_t)nt * 16 * DM + k1);

#pragma unroll
    for (int nt = 0; nt < 4; ++nt) {
      acc[0][nt] = wmma_bf16(a0c, bfc[nt], acc[0][nt]);
      acc[1][nt] = wmma_bf16(a1c, bfc[nt], acc[1][nt]);
    }

    const int k2 = kk + 64;
    if (k2 < DM) {
      a0c = ldfrag2(x0p + k2, x0p + k2 + 8);
      a1c = ldfrag2(x1p + k2, x1p + k2 + 8);
#pragma unroll
      for (int nt = 0; nt < 4; ++nt)
        bfc[nt] = ldfrag1(wr + (size_t)nt * 16 * DM + k2);
    }

#pragma unroll
    for (int nt = 0; nt < 4; ++nt) {
      acc[0][nt] = wmma_bf16(a0n, bfn[nt], acc[0][nt]);
      acc[1][nt] = wmma_bf16(a1n, bfn[nt], acc[1][nt]);
    }
  }

#pragma unroll
  for (int mt = 0; mt < 2; ++mt)
#pragma unroll
    for (int nt = 0; nt < 4; ++nt) {
      float bs = bO[n0 + nt * 16 + ln];
#pragma unroll
      for (int r = 0; r < 8; ++r)
        out[((size_t)b * SEQ + m0 + mt * 16 + half * 8 + r) * DM +
            n0 + nt * 16 + ln] = acc[mt][nt][r] + bs;
    }
}

// ---------------------------------------------------------------------------
extern "C" void kernel_launch(void* const* d_in, const int* in_sizes, int n_in,
                              void* d_out, int out_size, void* d_ws, size_t ws_size,
                              hipStream_t stream) {
  const float* x   = (const float*)d_in[0];
  const float* W_Q = (const float*)d_in[1];
  const float* W_K = (const float*)d_in[2];
  const float* W_V = (const float*)d_in[3];
  const float* W_O = (const float*)d_in[4];
  const float* b_Q = (const float*)d_in[5];
  const float* b_K = (const float*)d_in[6];
  const float* b_V = (const float*)d_in[7];
  const float* b_O = (const float*)d_in[8];
  float* out = (float*)d_out;

  const size_t NX = (size_t)BATCH * SEQ * DM;   // 4194304
  const size_t NW = (size_t)NH * DM * DH;       // 1048576

  __bf16* ws  = (__bf16*)d_ws;
  __bf16* Xb  = ws;
  __bf16* WqT = Xb  + NX;
  __bf16* WkT = WqT + NW;
  __bf16* WvT = WkT + NW;
  __bf16* WoT = WvT + NW;
  __bf16* Qb  = WoT + NW;
  __bf16* Kb  = Qb  + NX;
  __bf16* Vb  = Kb  + NX;
  __bf16* Zb  = Vb  + NX;

  cvt_f32_bf16<<<(int)(NX / 256), 256, 0, stream>>>(x, Xb, (int)NX);
  cvt_w_headT <<<(int)(NW / 256), 256, 0, stream>>>(W_Q, WqT);
  cvt_w_headT <<<(int)(NW / 256), 256, 0, stream>>>(W_K, WkT);
  cvt_w_headT <<<(int)(NW / 256), 256, 0, stream>>>(W_V, WvT);
  cvt_woT     <<<(int)(NW / 256), 256, 0, stream>>>(W_O, WoT);

  qkv_gemm<<<dim3(SEQ / 128, BATCH * NH, 3), 128, 0, stream>>>(
      Xb, WqT, WkT, WvT, b_Q, b_K, b_V, Qb, Kb, Vb);

  attn_kernel<<<dim3(SEQ / 128, BATCH * NH), 256, 0, stream>>>(Qb, Kb, Vb, Zb);

  out_gemm<<<dim3(SEQ / 128, DM / 64, BATCH), 128, 0, stream>>>(Zb, WoT, b_O, out);
}
